// Attention_77592879169835
// MI455X (gfx1250) — matrix operand order repaired and checked by reading, compiled, not run on hardware
//
#include <hip/hip_runtime.h>

typedef __bf16 bf16;
typedef __attribute__((ext_vector_type(16))) __bf16 bf16x16;
typedef __attribute__((ext_vector_type(8)))  __bf16 bf16x8;
typedef __attribute__((ext_vector_type(4)))  __bf16 bf16x4;
typedef __attribute__((ext_vector_type(8)))  float  f32x8;

#define B_        8
#define S_        4096
#define D_        64
#define NWAVE     8
#define RPW       16                 // query rows per wave (one 16-row A tile)
#define BR        (NWAVE * RPW)      // 128 rows per workgroup
#define BC        64                 // key tile
#define NITER     (S_ / BC)
#define NTHREADS  (NWAVE * 32)       // 256

union AFrag { bf16x16 v; bf16x8 h[2]; };

// ---- fused max+shuffle butterfly: one v_max_num_f32_dpp per stage ----
__device__ __forceinline__ float rmax16(float x) {   // max over 16-lane half
    asm("v_max_num_f32_dpp %0, %0, %0 row_xmask:1 row_mask:0xf bank_mask:0xf bound_ctrl:1" : "+v"(x));
    asm("v_max_num_f32_dpp %0, %0, %0 row_xmask:2 row_mask:0xf bank_mask:0xf bound_ctrl:1" : "+v"(x));
    asm("v_max_num_f32_dpp %0, %0, %0 row_xmask:4 row_mask:0xf bank_mask:0xf bound_ctrl:1" : "+v"(x));
    asm("v_max_num_f32_dpp %0, %0, %0 row_xmask:8 row_mask:0xf bank_mask:0xf bound_ctrl:1" : "+v"(x));
    return x;
}

__global__ __launch_bounds__(NTHREADS)
void fattn_wmma_bf16(const float* __restrict__ q,
                     const float* __restrict__ k,
                     const float* __restrict__ v,
                     const float* __restrict__ mask,
                     float* __restrict__ out)
{
    // double-buffered K / V^T tiles + per-wave P staging (48 KB total)
    __shared__ __align__(16) bf16 lds_k [2][BC * D_];       // [buf][key][d]
    __shared__ __align__(16) bf16 lds_vt[2][D_ * BC];       // [buf][d][key]
    __shared__ __align__(16) bf16 lds_p [NWAVE][RPW * BC];  // per-wave, private

    const int tid    = threadIdx.x;
    const int wave   = tid >> 5;
    const int lane   = tid & 31;
    const int lane16 = lane & 15;
    const int hlf    = lane >> 4;

    const int b      = blockIdx.x >> 5;      // 8 batches
    const int qTile  = blockIdx.x & 31;      // 32 q-tiles of 128 rows
    const int qBase  = qTile * BR + wave * RPW;

    // per-thread slot in the cooperative tile copy
    const int srow[4] = { (tid + 0*NTHREADS) >> 4, (tid + 1*NTHREADS) >> 4,
                          (tid + 2*NTHREADS) >> 4, (tid + 3*NTHREADS) >> 4 };
    const int sc4 = (tid & 15) << 2;         // d 0..60 (same for all 4 slots)

    // ---- Q fragment: two 16x32 bf16 A-frags, pre-scaled by 1/sqrt(D)*log2(e)
    const float qscale = 0.125f * 1.44269504088896340736f;
    AFrag qa[2];
    {
        const int kb = hlf * 16;
        const float* qrow = q + ((size_t)b * S_ + qBase + lane16) * D_;
#pragma unroll
        for (int kk = 0; kk < 2; ++kk)
#pragma unroll
            for (int i = 0; i < 8; ++i) {
                qa[kk].h[0][i] = (bf16)(qrow[kk * 32 + kb + i] * qscale);
                qa[kk].h[1][i] = (bf16)(qrow[kk * 32 + kb + 8 + i] * qscale);
            }
    }

    // ---- all-ones B fragment: P x ones == row-sum of P (softmax denom
    // accumulated as 5th output tile, same cf recurrence as O tiles)
    bf16x16 onesfrag;
#pragma unroll
    for (int i = 0; i < 16; ++i) onesfrag[i] = (bf16)1.0f;

    f32x8 acc[5];
#pragma unroll
    for (int n = 0; n < 5; ++n) acc[n] = (f32x8){};
    float mrow[8];
#pragma unroll
    for (int r = 0; r < 8; ++r) mrow[r] = -1e30f;

    const float* kg0 = k + (size_t)b * S_ * D_;
    const float* vg0 = v + (size_t)b * S_ * D_;

    // ---- prologue: stage tile 0 into buffer 0
#pragma unroll
    for (int it = 0; it < 4; ++it) {
        float4 kv4 = *(const float4*)(kg0 + srow[it] * D_ + sc4);
        bf16x4 kb4 = { (bf16)kv4.x, (bf16)kv4.y, (bf16)kv4.z, (bf16)kv4.w };
        *(bf16x4*)&lds_k[0][srow[it] * D_ + sc4] = kb4;
        float4 vv4 = *(const float4*)(vg0 + srow[it] * D_ + sc4);
        lds_vt[0][(sc4 + 0) * BC + srow[it]] = (bf16)vv4.x;
        lds_vt[0][(sc4 + 1) * BC + srow[it]] = (bf16)vv4.y;
        lds_vt[0][(sc4 + 2) * BC + srow[it]] = (bf16)vv4.z;
        lds_vt[0][(sc4 + 3) * BC + srow[it]] = (bf16)vv4.w;
    }
    __syncthreads();

    for (int t = 0; t < NITER; ++t) {
        const int cur = t & 1, nxt = cur ^ 1;

        // ---- issue NEXT tile's global loads first (latency hides under WMMAs)
        float4 kr[4], vr[4];
        if (t + 1 < NITER) {
            const float* kg = kg0 + (size_t)(t + 1) * BC * D_;
            const float* vg = vg0 + (size_t)(t + 1) * BC * D_;
#pragma unroll
            for (int it = 0; it < 4; ++it) {
                kr[it] = *(const float4*)(kg + srow[it] * D_ + sc4);
                vr[it] = *(const float4*)(vg + srow[it] * D_ + sc4);
            }
        }
        if (t + 2 < NITER) {                    // gfx1250 global_prefetch_b8
            __builtin_prefetch(kg0 + (size_t)(t + 2) * BC * D_ + tid * 4, 0, 3);
            __builtin_prefetch(vg0 + (size_t)(t + 2) * BC * D_ + tid * 4, 0, 3);
        }

        // ---- GEMM1: S = Q * K^T from buffer `cur`
        f32x8 sc[4];
#pragma unroll
        for (int n = 0; n < 4; ++n) sc[n] = (f32x8){};
#pragma unroll
        for (int kk = 0; kk < 2; ++kk)
#pragma unroll
            for (int n = 0; n < 4; ++n) {
                const int key = n * 16 + lane16;
                bf16x16 kfrag = *(const bf16x16*)&lds_k[cur][key * D_ + kk * 32 + hlf * 16];
                sc[n] = __builtin_amdgcn_wmma_f32_16x16x32_bf16(
                            false, qa[kk].v, false, kfrag, (short)0, sc[n], false, false);
            }

        // ---- online softmax: fused-DPP max butterfly; sum via acc[4] WMMA
#pragma unroll
        for (int r = 0; r < 8; ++r) {
            float mx = fmaxf(fmaxf(sc[0][r], sc[1][r]),
                             fmaxf(sc[2][r], sc[3][r]));
            const float mnew = fmaxf(mrow[r], rmax16(mx));
            const float cf = __builtin_amdgcn_exp2f(mrow[r] - mnew);
            mrow[r] = mnew;
#pragma unroll
            for (int n = 0; n < 4; ++n)
                sc[n][r] = __builtin_amdgcn_exp2f(sc[n][r] - mnew);
#pragma unroll
            for (int n = 0; n < 5; ++n)
                acc[n][r] *= cf;
        }

        // ---- stage P (wave-private slab; HW keeps a wave's DS ops in order,
        //      only need a compiler barrier, not a workgroup barrier)
#pragma unroll
        for (int n = 0; n < 4; ++n)
#pragma unroll
            for (int r = 0; r < 8; ++r)
                lds_p[wave][(r + 8 * hlf) * BC + n * 16 + lane16] = (bf16)sc[n][r];
        asm volatile("" ::: "memory");

        // ---- GEMM2: O += P * V (+ row-sum tile) from buffer `cur`
        AFrag pa[2];
        {
            const int kb = hlf * 16;
#pragma unroll
            for (int kk = 0; kk < 2; ++kk) {
                const bf16* base = &lds_p[wave][lane16 * BC + kk * 32];
                pa[kk].h[0] = *(const bf16x8*)(base + kb);
                pa[kk].h[1] = *(const bf16x8*)(base + kb + 8);
            }
        }
#pragma unroll
        for (int kk = 0; kk < 2; ++kk) {
#pragma unroll
            for (int n = 0; n < 4; ++n) {
                const int d = n * 16 + lane16;
                bf16x16 vfrag = *(const bf16x16*)&lds_vt[cur][d * BC + kk * 32 + hlf * 16];
                acc[n] = __builtin_amdgcn_wmma_f32_16x16x32_bf16(
                             false, pa[kk].v, false, vfrag, (short)0, acc[n], false, false);
            }
            acc[4] = __builtin_amdgcn_wmma_f32_16x16x32_bf16(
                         false, pa[kk].v, false, onesfrag, (short)0, acc[4], false, false);
        }

        // ---- convert + store NEXT tile into buffer `nxt`
        if (t + 1 < NITER) {
#pragma unroll
            for (int it = 0; it < 4; ++it) {
                bf16x4 kb4 = { (bf16)kr[it].x, (bf16)kr[it].y,
                               (bf16)kr[it].z, (bf16)kr[it].w };
                *(bf16x4*)&lds_k[nxt][srow[it] * D_ + sc4] = kb4;
                lds_vt[nxt][(sc4 + 0) * BC + srow[it]] = (bf16)vr[it].x;
                lds_vt[nxt][(sc4 + 1) * BC + srow[it]] = (bf16)vr[it].y;
                lds_vt[nxt][(sc4 + 2) * BC + srow[it]] = (bf16)vr[it].z;
                lds_vt[nxt][(sc4 + 3) * BC + srow[it]] = (bf16)vr[it].w;
            }
        }
        __syncthreads();        // single barrier per iteration
    }

    // ---- epilogue: normalize by acc[4] (= softmax denom), mask, store fp32
#pragma unroll
    for (int r = 0; r < 8; ++r) {
        const int rowg = qBase + r + 8 * hlf;
        const float s = mask[(size_t)b * S_ + rowg] / acc[4][r];
        float* orow = out + ((size_t)b * S_ + rowg) * D_;
#pragma unroll
        for (int n = 0; n < 4; ++n)
            orow[n * 16 + lane16] = acc[n][r] * s;
    }
}

extern "C" void kernel_launch(void* const* d_in, const int* in_sizes, int n_in,
                              void* d_out, int out_size, void* d_ws, size_t ws_size,
                              hipStream_t stream) {
    const float* q    = (const float*)d_in[0];
    const float* k    = (const float*)d_in[1];
    const float* v    = (const float*)d_in[2];
    const float* mask = (const float*)d_in[3];
    float* out = (float*)d_out;
    dim3 grid(B_ * (S_ / BR));     // 256 workgroups
    dim3 block(NTHREADS);          // 8 wave32 per WG
    fattn_wmma_bf16<<<grid, block, 0, stream>>>(q, k, v, mask, out);
}
